// Mul_Attention_83451214561717
// MI455X (gfx1250) — matrix operand order repaired and checked by reading, not compiled, not run on hardware
//
#include <hip/hip_runtime.h>
#include <hip/hip_bf16.h>

// ---------------------------------------------------------------------------
// Problem constants (reference: B=4, S=2048, H=1024)
// ---------------------------------------------------------------------------
constexpr int HH = 1024;
constexpr int SS = 2048;
constexpr int BB = 4;
constexpr int MTOT = BB * SS; // 8192 rows for the projection GEMMs

typedef __attribute__((ext_vector_type(16))) __bf16        v16bf;
typedef __attribute__((ext_vector_type(8)))  float         v8f;
typedef __attribute__((ext_vector_type(8)))  unsigned int  v8u;

// ---------------------------------------------------------------------------
// bf16 helpers (round-to-nearest-even), no __bf16 arithmetic needed
// ---------------------------------------------------------------------------
__device__ __forceinline__ unsigned short f32_to_bf16(float f) {
  unsigned int u = __builtin_bit_cast(unsigned int, f);
  unsigned int r = u + 0x7fffu + ((u >> 16) & 1u);
  return (unsigned short)(r >> 16);
}
__device__ __forceinline__ float bf16_to_f32(unsigned short h) {
  unsigned int u = ((unsigned int)h) << 16;
  return __builtin_bit_cast(float, u);
}
__device__ __forceinline__ unsigned int pack2(unsigned short lo, unsigned short hi) {
  return (unsigned int)lo | ((unsigned int)hi << 16);
}

// A-operand k order used in this file (contiguous; AMD's table order differs): VGPR j holds a K-pair.
// lanes 0-15 : K=2j      (K 0..15)
// lanes16-31 : K=2j+16   (K 16..31)
__device__ __forceinline__ int a_pairK(int j, int hi16) {
  return 2 * j + (hi16 ? 16 : 0);
}

__device__ __forceinline__ v8f wmma_bf16(v16bf a, v16bf b, v8f c) {
  return __builtin_amdgcn_wmma_f32_16x16x32_bf16(false, a, false, b,
                                                 (short)0, c, false, false);
}

// Async global->LDS copy of 16 bytes per lane (gfx1250 GVS form), ASYNCcnt.
__device__ __forceinline__ void async_b128(unsigned lds_off, unsigned g_off,
                                           const void* base) {
  asm volatile("global_load_async_to_lds_b128 %0, %1, %2"
               :: "v"(lds_off), "v"(g_off), "s"(base) : "memory");
}

// ---------------------------------------------------------------------------
// Kernel 1: ReLU + f32->bf16 of the input activations
// ---------------------------------------------------------------------------
__global__ __launch_bounds__(256)
void cvt_x_kernel(const float* __restrict__ x, unsigned short* __restrict__ xb) {
  int idx = blockIdx.x * blockDim.x + threadIdx.x;
  if (idx < MTOT * HH) {
    float v = x[idx];
    v = v > 0.0f ? v : 0.0f;
    xb[idx] = f32_to_bf16(v);
  }
}

// ---------------------------------------------------------------------------
// Kernel 2: weights f32->bf16 (Wk then Wq)
// ---------------------------------------------------------------------------
__global__ __launch_bounds__(256)
void cvt_w_kernel(const float* __restrict__ Wk, const float* __restrict__ Wq,
                  unsigned short* __restrict__ Wkb, unsigned short* __restrict__ Wqb) {
  int idx = blockIdx.x * blockDim.x + threadIdx.x;
  constexpr int N = HH * HH;
  if (idx < N)          Wkb[idx]     = f32_to_bf16(Wk[idx]);
  else if (idx < 2 * N) Wqb[idx - N] = f32_to_bf16(Wq[idx - N]);
}

// ---------------------------------------------------------------------------
// Kernel 3: projection GEMMs   out[m,o] = relu(x)[m,:] . W[o,:] + b[o]
//
// Block = 256 threads (8 waves) -> 128x128 output tile.
// Waves in a 4(M) x 2(N) grid: each wave computes 32x64 = 8 WMMA tiles.
// K-loop in 32-wide chunks, double-buffered LDS staging filled with
// global_load_async_to_lds_b128 (ASYNCcnt), WMMA fed from LDS.
//   A tile: 128 x 32 bf16 (8 KB) -> 512 x 16B slots (2 per thread)
//   B tile: 128 x 32 bf16 (8 KB) -> 512 x 16B slots (2 per thread)
// blockIdx.y: 0 => K projection, 1 => Q projection
// ---------------------------------------------------------------------------
__global__ __launch_bounds__(256)
void proj_gemm_kernel(const unsigned short* __restrict__ xb,
                      const unsigned short* __restrict__ Wkb,
                      const unsigned short* __restrict__ Wqb,
                      const float* __restrict__ bk,
                      const float* __restrict__ bq,
                      unsigned short* __restrict__ Kb,
                      unsigned short* __restrict__ Qb) {
  __shared__ unsigned short shA[2][128][32];  // [buf][row][k]
  __shared__ unsigned short shB[2][128][32];  // [buf][n][k]

  const int tid  = threadIdx.x;
  const int lane = tid & 31;
  const int hi16 = (lane >> 4) & 1;
  const int lml  = lane & 15;
  const int wv   = tid >> 5;                  // wave 0..7
  const int mw   = (wv & 3) * 32;             // wave M base within block
  const int nw   = (wv >> 2) * 64;            // wave N base within block

  const int blk  = blockIdx.x;                // 0..511
  const int mBlk = (blk >> 3) * 128;          // 64 m-blocks of 128 rows
  const int n0   = (blk & 7) * 128;           // 8 n-groups of 128 cols

  const unsigned short* W  = blockIdx.y ? Wqb : Wkb;
  const float*         bia = blockIdx.y ? bq  : bk;
  unsigned short*      Out = blockIdx.y ? Qb  : Kb;

  // per-thread async slot geometry: slot s -> row = s>>2, 16B segment = s&3
  const int r0 = tid >> 2,         sg0 = tid & 3;           // slot tid
  const int r1 = (tid + 256) >> 2, sg1 = tid & 3;           // slot tid+256

  const unsigned ldsA = (unsigned)(uintptr_t)(&shA[0][0][0]);
  const unsigned ldsB = (unsigned)(uintptr_t)(&shB[0][0][0]);

  auto issue_stage = [&](int buf, int k0) {
    const unsigned bufA = ldsA + (unsigned)buf * (128u * 64u);
    const unsigned bufB = ldsB + (unsigned)buf * (128u * 64u);
    async_b128(bufA + (unsigned)tid * 16u,
               (unsigned)(((mBlk + r0) * HH + k0) * 2 + sg0 * 16), xb);
    async_b128(bufA + (unsigned)(tid + 256) * 16u,
               (unsigned)(((mBlk + r1) * HH + k0) * 2 + sg1 * 16), xb);
    async_b128(bufB + (unsigned)tid * 16u,
               (unsigned)(((n0 + r0) * HH + k0) * 2 + sg0 * 16), W);
    async_b128(bufB + (unsigned)(tid + 256) * 16u,
               (unsigned)(((n0 + r1) * HH + k0) * 2 + sg1 * 16), W);
  };

  v8f acc[2][4] = {};
  issue_stage(0, 0);

  for (int ks = 0; ks < 32; ++ks) {
    const int buf = ks & 1;
    if (ks + 1 < 32) {
      issue_stage(buf ^ 1, (ks + 1) * 32);
      asm volatile("s_wait_asynccnt 0x4" ::: "memory");  // stage ks landed
    } else {
      asm volatile("s_wait_asynccnt 0x0" ::: "memory");
    }
    __syncthreads();

    // Load ALL fragments first so WMMAs can overlap the tail DS loads.
    v8u au[2], bu[4];
#pragma unroll
    for (int mi = 0; mi < 2; ++mi) {
      const unsigned short* arow = &shA[buf][mw + mi * 16 + lml][0];
#pragma unroll
      for (int j = 0; j < 8; ++j)
        au[mi][j] = *(const unsigned int*)(arow + a_pairK(j, hi16));
    }
#pragma unroll
    for (int n = 0; n < 4; ++n) {
      const unsigned short* brow = &shB[buf][nw + n * 16 + lml][hi16 * 16];
#pragma unroll
      for (int j = 0; j < 8; ++j)
        bu[n][j] = *(const unsigned int*)(brow + 2 * j);
    }
#pragma unroll
    for (int mi = 0; mi < 2; ++mi) {
      v16bf a = __builtin_bit_cast(v16bf, au[mi]);
#pragma unroll
      for (int n = 0; n < 4; ++n)
        acc[mi][n] = wmma_bf16(a, __builtin_bit_cast(v16bf, bu[n]), acc[mi][n]);
    }
    __syncthreads();
  }

  // epilogue: +bias, cvt to bf16, store (C/D layout: lane<16 -> M=v, N=lane)
#pragma unroll
  for (int mi = 0; mi < 2; ++mi) {
    const int m0 = mBlk + mw + mi * 16;
#pragma unroll
    for (int n = 0; n < 4; ++n) {
      const int ncol = n0 + nw + n * 16 + lml;
      const float bs = bia[ncol];
#pragma unroll
      for (int v = 0; v < 8; ++v) {
        const int mrow = m0 + v + hi16 * 8;
        Out[(size_t)mrow * HH + ncol] = f32_to_bf16(acc[mi][n][v] + bs);
      }
    }
  }
}

// ---------------------------------------------------------------------------
// Kernel 4: per-batch column mean of K (for uniform-softmax batches)
// ---------------------------------------------------------------------------
__global__ __launch_bounds__(256)
void col_mean_kernel(const unsigned short* __restrict__ Kb, float* __restrict__ means) {
  int idx = blockIdx.x * blockDim.x + threadIdx.x;
  if (idx >= BB * HH) return;
  const int b = idx / HH;
  const int h = idx - b * HH;
  const unsigned short* p = Kb + (size_t)b * SS * HH + h;
  float s = 0.0f;
  for (int r = 0; r < SS; ++r) s += bf16_to_f32(p[(size_t)r * HH]);
  means[idx] = s * (1.0f / (float)SS);
}

// ---------------------------------------------------------------------------
// Kernel 5: p_attn fill — batch0 plane = 0 (band overwritten later),
//           batches 1..3 = uniform 1/S
// ---------------------------------------------------------------------------
__global__ __launch_bounds__(256)
void fill_pattn_kernel(float* __restrict__ pattn) {
  size_t idx = (size_t)blockIdx.x * blockDim.x + threadIdx.x;
  constexpr size_t NTOT = (size_t)BB * SS * SS;
  constexpr size_t PLANE = (size_t)SS * SS;
  if (idx < NTOT)
    pattn[idx] = (idx < PLANE) ? 0.0f : (1.0f / (float)SS);
}

// ---------------------------------------------------------------------------
// Kernel 6: out for batches 1..3 = broadcast column mean
// ---------------------------------------------------------------------------
__global__ __launch_bounds__(256)
void bcast_out_kernel(const float* __restrict__ means, float* __restrict__ out) {
  size_t idx = (size_t)blockIdx.x * blockDim.x + threadIdx.x;
  constexpr size_t N = (size_t)3 * SS * HH;  // batches 1..3
  if (idx >= N) return;
  const size_t plane = (size_t)SS * HH;
  const int b = 1 + (int)(idx / plane);
  const int h = (int)(idx % (size_t)HH);
  out[plane + idx] = means[b * HH + h];
}

// ---------------------------------------------------------------------------
// Kernel 7: banded attention for batch 0.
//   One wave per 16-query tile (single-wave WG: barriers are S_NOP).
//   5 key tiles of 16 cover the band [i0-64, i0+15].
//   scores via WMMA (Q.K^T), masked softmax in LDS, p_attn band write,
//   then out = P.K via WMMA over 3 K-chunks x 64 N-tiles.
// ---------------------------------------------------------------------------
__global__ __launch_bounds__(32)
void band_attn_kernel(const unsigned short* __restrict__ Qb,
                      const unsigned short* __restrict__ Kb,
                      const int* __restrict__ kmul_p,
                      float* __restrict__ out,
                      float* __restrict__ pattn) {
  __shared__ float sc[16][80];   // masked scores
  __shared__ float pb[16][96];   // probabilities, zero-padded to K=96

  const int lane = threadIdx.x;
  const int hi16 = (lane >> 4) & 1;
  const int lml  = lane & 15;
  const int i0   = blockIdx.x * 16;
  int kw = *kmul_p;
  if (kw > 64) kw = 64;
  if (kw < 0)  kw = 0;
  const int jbase = i0 - 64;

  for (int t = lane; t < 16 * 80; t += 32) ((float*)sc)[t] = -3.0e38f;
  for (int t = lane; t < 16 * 96; t += 32) ((float*)pb)[t] = 0.0f;
  __syncthreads();

  // ---- scores: 5 tiles of QK^T, scaled by 1/sqrt(H) = 1/32 ----
  for (int t = 0; t < 5; ++t) {
    const int jt = jbase + t * 16;
    if (jt < 0) continue;
    v8f acc = {};
    const unsigned short* qrow = Qb + (size_t)(i0 + lml) * HH;          // batch 0
    const unsigned short* krow = Kb + (size_t)(jt + lml) * HH + hi16 * 16;
    for (int k0 = 0; k0 < HH; k0 += 32) {
      v8u au, bu;
#pragma unroll
      for (int j = 0; j < 8; ++j) {
        au[j] = *(const unsigned int*)(qrow + k0 + a_pairK(j, hi16));
        bu[j] = *(const unsigned int*)(krow + k0 + 2 * j);
      }
      acc = wmma_bf16(__builtin_bit_cast(v16bf, au),
                      __builtin_bit_cast(v16bf, bu), acc);
    }
#pragma unroll
    for (int v = 0; v < 8; ++v)
      sc[v + hi16 * 8][t * 16 + lml] = acc[v] * 0.03125f;
  }
  __syncthreads();

  // ---- masked softmax, one lane per query row; write p_attn band ----
  if (lane < 16) {
    const int i = i0 + lane;
    float mx = -3.0e38f;
    for (int c = 0; c < 80; ++c) {
      const int jg = jbase + c;
      const bool valid = (jg >= 0) && (jg <= i) && (jg >= i - kw);
      const float s = valid ? sc[lane][c] : -3.0e38f;
      sc[lane][c] = s;
      if (s > mx) mx = s;
    }
    float sum = 0.0f;
    for (int c = 0; c < 80; ++c) {
      const float s = sc[lane][c];
      const float e = (s <= -1.0e38f) ? 0.0f : __expf(s - mx);
      pb[lane][c] = e;
      sum += e;
    }
    const float inv = 1.0f / sum;
    for (int c = 0; c < 80; ++c) {
      const float p = pb[lane][c] * inv;
      pb[lane][c] = p;
      const int jg = jbase + c;
      if (jg >= 0) pattn[(size_t)i * SS + jg] = p;
    }
  }
  __syncthreads();

  // ---- out = P (16x96) . K (96xH) via WMMA, 3 K-chunks ----
  v16bf afr[3];
#pragma unroll
  for (int kk = 0; kk < 3; ++kk) {
    v8u au;
#pragma unroll
    for (int j = 0; j < 8; ++j) {
      const int K0 = kk * 32 + a_pairK(j, hi16);
      au[j] = pack2(f32_to_bf16(pb[lml][K0]), f32_to_bf16(pb[lml][K0 + 1]));
    }
    afr[kk] = __builtin_bit_cast(v16bf, au);
  }

  for (int h0 = 0; h0 < HH; h0 += 16) {
    v8f acc = {};
#pragma unroll
    for (int kk = 0; kk < 3; ++kk) {
      v8u bu;
#pragma unroll
      for (int j = 0; j < 8; ++j) {
        int k0r = kk * 32 + hi16 * 16 + 2 * j;
        int j0r = jbase + k0r;
        int j1r = j0r + 1;
        j0r = j0r < 0 ? 0 : (j0r >= SS ? SS - 1 : j0r);
        j1r = j1r < 0 ? 0 : (j1r >= SS ? SS - 1 : j1r);
        bu[j] = pack2(Kb[(size_t)j0r * HH + h0 + lml],
                      Kb[(size_t)j1r * HH + h0 + lml]);
      }
      acc = wmma_bf16(afr[kk], __builtin_bit_cast(v16bf, bu), acc);
    }
#pragma unroll
    for (int v = 0; v < 8; ++v)
      out[(size_t)(i0 + v + hi16 * 8) * HH + h0 + lml] = acc[v];
  }
}

// ---------------------------------------------------------------------------
// Host launcher
// ---------------------------------------------------------------------------
extern "C" void kernel_launch(void* const* d_in, const int* in_sizes, int n_in,
                              void* d_out, int out_size, void* d_ws, size_t ws_size,
                              hipStream_t stream) {
  (void)in_sizes; (void)n_in; (void)out_size; (void)ws_size;

  const float* x    = (const float*)d_in[0];
  const float* Wk   = (const float*)d_in[1];
  const float* bk   = (const float*)d_in[2];
  const float* Wq   = (const float*)d_in[3];
  const float* bq   = (const float*)d_in[4];
  const int*   kmul = (const int*)d_in[5];

  // workspace layout (bytes)
  char* ws = (char*)d_ws;
  unsigned short* xb    = (unsigned short*)(ws);                 // 16 MB
  unsigned short* Wkb   = (unsigned short*)(ws + 16777216);      //  2 MB
  unsigned short* Wqb   = (unsigned short*)(ws + 18874368);      //  2 MB
  unsigned short* Kb    = (unsigned short*)(ws + 20971520);      // 16 MB
  unsigned short* Qb    = (unsigned short*)(ws + 37748736);      // 16 MB
  float*          means = (float*)(ws + 54525952);               // 16 KB

  float* out   = (float*)d_out;
  float* pattn = out + (size_t)BB * SS * HH;

  cvt_x_kernel<<<(MTOT * HH + 255) / 256, 256, 0, stream>>>(x, xb);
  cvt_w_kernel<<<(2 * HH * HH + 255) / 256, 256, 0, stream>>>(Wk, Wq, Wkb, Wqb);
  proj_gemm_kernel<<<dim3(512, 2), 256, 0, stream>>>(xb, Wkb, Wqb, bk, bq, Kb, Qb);
  col_mean_kernel<<<(BB * HH + 255) / 256, 256, 0, stream>>>(Kb, means);
  fill_pattn_kernel<<<(int)(((size_t)BB * SS * SS + 255) / 256), 256, 0, stream>>>(pattn);
  bcast_out_kernel<<<(int)(((size_t)3 * SS * HH + 255) / 256), 256, 0, stream>>>(means, out);
  band_attn_kernel<<<SS / 16, 32, 0, stream>>>(Qb, Kb, kmul, out, pattn);
}
